// TFGPTJAttention_68264210202909
// MI455X (gfx1250) — matrix operand order repaired and checked by reading, not compiled, not run on hardware
//
#include <hip/hip_runtime.h>
#include <stdint.h>

typedef unsigned short u16;
typedef __attribute__((ext_vector_type(16))) __bf16 v16bf;
typedef __attribute__((ext_vector_type(8)))  float  v8f;

#define D_MODEL 4096
#define SEQ     2048
#define NHEAD   16
#define HDIM    256

__device__ __forceinline__ u16 f2bf(float f) {
  unsigned int u = __float_as_uint(f);
  u += 0x7FFFu + ((u >> 16) & 1u);          // round-to-nearest-even
  return (u16)(u >> 16);
}
__device__ __forceinline__ float bf2f(u16 h) {
  return __uint_as_float(((unsigned int)h) << 16);
}

union FragBF { v16bf v; uint4 q[2]; u16 s[16]; };

__device__ __forceinline__ v8f vzero8() {
  v8f v;
#pragma unroll
  for (int i = 0; i < 8; ++i) v[i] = 0.f;
  return v;
}

// ---------------------------------------------------------------------------
// f32 -> bf16 bulk conversion (row-major preserved; used for X)
// ---------------------------------------------------------------------------
__global__ __launch_bounds__(256)
void cvt_f32_bf16_k(const float* __restrict__ src, u16* __restrict__ dst, int n) {
  int i = (blockIdx.x * 256 + threadIdx.x) * 4;
  if (i >= n) return;
  float4 f = *(const float4*)(src + i);
  uint2 o;
  o.x = (unsigned int)f2bf(f.x) | ((unsigned int)f2bf(f.y) << 16);
  o.y = (unsigned int)f2bf(f.z) | ((unsigned int)f2bf(f.w) << 16);
  *(uint2*)(dst + i) = o;
}

// ---------------------------------------------------------------------------
// f32 -> bf16 conversion WITH transpose: W[R][C] f32 -> WT[C][R] bf16.
// One-time pass; folds the GEMM's B-side transpose out of the hot loop so the
// main GEMM needs no LDS at all. 32x32 LDS tile, 256 threads.
// ---------------------------------------------------------------------------
__global__ __launch_bounds__(256)
void cvtT_f32_bf16_k(const float* __restrict__ src, u16* __restrict__ dst,
                     int R, int C) {
  __shared__ float tile[32][33];
  const int tx = threadIdx.x & 31;
  const int ty = threadIdx.x >> 5;            // 0..7
  const int r0 = blockIdx.y * 32;
  const int c0 = blockIdx.x * 32;
#pragma unroll
  for (int i = 0; i < 32; i += 8)
    tile[ty + i][tx] = src[(size_t)(r0 + ty + i) * C + c0 + tx];
  __syncthreads();
#pragma unroll
  for (int i = 0; i < 32; i += 8)
    dst[(size_t)(c0 + ty + i) * R + r0 + tx] = f2bf(tile[tx][ty + i]);
}

// ---------------------------------------------------------------------------
// bf16 N-T GEMM: C[M,N] = A[M,K] * BT[N,K]^T, f32 accumulate via WMMA.
// Both operand fragments are contiguous 16B per-lane global loads (CDNA5
// wave32 WMMA A/B layouts), so there is no LDS staging and no barrier in the
// K loop -> compiler emits a pure clause'd load / v_wmma pipeline.
// 256 threads = 8 waves, WG tile 128x128, wave tile 32x64, BK = 32.
// ---------------------------------------------------------------------------
template <bool OUT_F32>
__global__ __launch_bounds__(256, 1)
void gemm_nt_k(const u16* __restrict__ A, const u16* __restrict__ BT,
               void* __restrict__ Cout, int M, int N, int K) {
  const int lane = threadIdx.x & 31;
  const int wid  = threadIdx.x >> 5;
  const int wr   = wid & 3;                   // wave row group (4 x 32 rows)
  const int wc   = wid >> 2;                  // wave col group (2 x 64 cols)
  const int half = lane >> 4;
  const int l16  = lane & 15;
  const int blockM = blockIdx.y * 128;
  const int blockN = blockIdx.x * 128;

  v8f acc[2][4];
#pragma unroll
  for (int mt = 0; mt < 2; ++mt)
#pragma unroll
    for (int nt = 0; nt < 4; ++nt) acc[mt][nt] = vzero8();

  const u16* ap0 = A + (size_t)(blockM + wr * 32 + l16) * K;
  const u16* ap1 = ap0 + (size_t)16 * K;
  const u16* bp[4];
#pragma unroll
  for (int nt = 0; nt < 4; ++nt)
    bp[nt] = BT + (size_t)(blockN + wc * 64 + nt * 16 + l16) * K;

  for (int kk = 0; kk < K; kk += 32) {
    FragBF af[2];
    af[0].q[0] = *(const uint4*)(ap0 + kk + half * 8);
    af[0].q[1] = *(const uint4*)(ap0 + kk + 16 + half * 8);
    af[1].q[0] = *(const uint4*)(ap1 + kk + half * 8);
    af[1].q[1] = *(const uint4*)(ap1 + kk + 16 + half * 8);
#pragma unroll
    for (int nt = 0; nt < 4; ++nt) {
      FragBF bf;                              // elem e<8: B[kk+half*8+e][n], e>=8: B[kk+16+half*8+e-8][n]
      bf.q[0] = *(const uint4*)(bp[nt] + kk + half * 8);
      bf.q[1] = *(const uint4*)(bp[nt] + kk + 16 + half * 8);
#pragma unroll
      for (int mt = 0; mt < 2; ++mt)
        acc[mt][nt] = __builtin_amdgcn_wmma_f32_16x16x32_bf16(
            false, af[mt].v, false, bf.v, (short)0, acc[mt][nt], false, false);
    }
  }

#pragma unroll
  for (int mt = 0; mt < 2; ++mt)
#pragma unroll
    for (int nt = 0; nt < 4; ++nt) {
      int col = blockN + wc * 64 + nt * 16 + l16;
#pragma unroll
      for (int r = 0; r < 8; ++r) {
        int row = blockM + wr * 32 + mt * 16 + half * 8 + r;
        if (OUT_F32)
          ((float*)Cout)[(size_t)row * N + col] = acc[mt][nt][r];
        else
          ((u16*)Cout)[(size_t)row * N + col] = f2bf(acc[mt][nt][r]);
      }
    }
}

// ---------------------------------------------------------------------------
// GPT-J rotary (rotate_every_two) on first 64 dims of each head, in-place bf16.
// ---------------------------------------------------------------------------
__global__ __launch_bounds__(256)
void rotary_k(u16* __restrict__ Q, u16* __restrict__ K, const int* __restrict__ pos) {
  int t = blockIdx.x * 256 + threadIdx.x;     // 0 .. SEQ*NHEAD*32-1
  int i = t & 31;                             // frequency index
  int h = (t >> 5) & 15;
  int s = t >> 9;
  float theta = (float)pos[s] * __powf(10000.f, -((float)(2 * i)) * (1.f / 64.f));
  float sn = __sinf(theta), cs = __cosf(theta);
  size_t base = (size_t)s * D_MODEL + h * HDIM + 2 * i;
  {
    float a = bf2f(Q[base]), b = bf2f(Q[base + 1]);
    Q[base]     = f2bf(a * cs - b * sn);
    Q[base + 1] = f2bf(b * cs + a * sn);
  }
  {
    float a = bf2f(K[base]), b = bf2f(K[base + 1]);
    K[base]     = f2bf(a * cs - b * sn);
    K[base + 1] = f2bf(b * cs + a * sn);
  }
}

// ---------------------------------------------------------------------------
// Causal flash attention, bf16 QKV (token-major [S, 4096]), f32 online softmax.
// 1 WG (4 waves, 128 thr) per (64-query block, head). Wave owns 16 q rows.
// Q fragments cached in registers across the key loop.
// Per key block:
//   - K tile (64x256, row-major, needs NO transpose for QK^T B-fragments) is
//     staged into LDS with CDNA5 global_load_async_to_lds_b128 (ASYNCcnt),
//     overlapped with the manual V-tile transpose staging (VALU+ds work),
//     then s_wait_asynccnt 0 + workgroup barrier.
//   - P round-trips through per-wave LDS to reshape C-layout -> A-layout.
// ---------------------------------------------------------------------------
__global__ __launch_bounds__(128, 1)
void flash_attn_k(const u16* __restrict__ Q, const u16* __restrict__ K,
                  const u16* __restrict__ V, u16* __restrict__ O) {
  __shared__ u16 kT[64][256];                 // [key][dim]   (async-staged)
  __shared__ u16 vT[HDIM][72];                // [vcol][key]  (manual transpose)
  __shared__ u16 pT[4][16][72];               // [wave][qrow][key]
  const int tid  = threadIdx.x;
  const int lane = tid & 31;
  const int w    = tid >> 5;
  const int half = lane >> 4;
  const int l16  = lane & 15;
  const int qb   = blockIdx.x;                // 0..31
  const int h    = blockIdx.y;                // 0..15
  const int qBase = qb * 64;
  const int qRow0 = qBase + w * 16;
  const size_t colQ = (size_t)h * HDIM;

  // cache Q fragments: 8 chunks of K-dim (256 = 8 x 32)
  FragBF qf[8];
  {
    const u16* qp = Q + (size_t)(qRow0 + l16) * D_MODEL + colQ;
#pragma unroll
    for (int c = 0; c < 8; ++c) {
      qf[c].q[0] = *(const uint4*)(qp + c * 32 + half * 8);
      qf[c].q[1] = *(const uint4*)(qp + c * 32 + 16 + half * 8);
    }
  }

  v8f acc[16];
#pragma unroll
  for (int t2 = 0; t2 < 16; ++t2) acc[t2] = vzero8();
  float Mrun[8], Lrun[8];
#pragma unroll
  for (int r = 0; r < 8; ++r) { Mrun[r] = -3.0e38f; Lrun[r] = 0.f; }

  const int qRowBase = qRow0 + half * 8;      // global q row of acc element r=0

  for (int kb = 0; kb <= qb; ++kb) {
    const int keyBase = kb * 64;
    __syncthreads();                          // previous iter's LDS readers done

    // ---- async-copy K tile into LDS (16B per lane per issue, ASYNCcnt) ----
#pragma unroll
    for (int s2 = 0; s2 < 16; ++s2) {
      int slot = s2 * 128 + tid;              // 0..2047
      int key  = slot >> 5;
      int cg   = slot & 31;
      unsigned long long ga =
          (unsigned long long)(const void*)(K + (size_t)(keyBase + key) * D_MODEL + colQ + cg * 8);
      unsigned la = (unsigned)(size_t)(const void*)&kT[key][cg * 8];
      asm volatile("global_load_async_to_lds_b128 %0, %1, off"
                   :: "v"(la), "v"(ga) : "memory");
    }

    // ---- meanwhile: stage V tile transposed -> vT[vcol][key] ----
#pragma unroll
    for (int s2 = 0; s2 < 16; ++s2) {
      int slot = s2 * 128 + tid;              // 0..2047
      int key  = slot >> 5;
      int cg   = slot & 31;
      uint4 vv = *(const uint4*)(V + (size_t)(keyBase + key) * D_MODEL + colQ + cg * 8);
      const u16* pv = (const u16*)&vv;
#pragma unroll
      for (int j = 0; j < 8; ++j) vT[cg * 8 + j][key] = pv[j];
    }

    asm volatile("s_wait_asynccnt 0x0" ::: "memory");
    __syncthreads();

    // ---- scores S = Q * K^T (16 x 64 per wave), K fragments from LDS ----
    v8f st[4];
#pragma unroll
    for (int j = 0; j < 4; ++j) {
      st[j] = vzero8();
      const u16* kp = &kT[j * 16 + l16][0];
#pragma unroll
      for (int c = 0; c < 8; ++c) {
        FragBF kf;
        kf.q[0] = *(const uint4*)(kp + c * 32 + half * 8);
        kf.q[1] = *(const uint4*)(kp + c * 32 + 16 + half * 8);
        st[j] = __builtin_amdgcn_wmma_f32_16x16x32_bf16(
            false, qf[c].v, false, kf.v, (short)0, st[j], false, false);
      }
    }

    // ---- causal mask + 1/sqrt(256) scale ----
#pragma unroll
    for (int j = 0; j < 4; ++j) {
      int kg = keyBase + j * 16 + l16;
#pragma unroll
      for (int r = 0; r < 8; ++r) {
        float v = st[j][r] * 0.0625f;
        st[j][r] = (kg > qRowBase + r) ? -6.25e7f : v;  // MASK_BIAS / 16
      }
    }

    // ---- online softmax (row stats over the 16 lanes of each half) ----
#pragma unroll
    for (int r = 0; r < 8; ++r) {
      float rm = fmaxf(fmaxf(st[0][r], st[1][r]), fmaxf(st[2][r], st[3][r]));
#pragma unroll
      for (int m2 = 1; m2 < 16; m2 <<= 1) rm = fmaxf(rm, __shfl_xor(rm, m2, 32));
      float nM    = fmaxf(Mrun[r], rm);
      float alpha = __expf(Mrun[r] - nM);
      float rs = 0.f;
#pragma unroll
      for (int j = 0; j < 4; ++j) { float p = __expf(st[j][r] - nM); st[j][r] = p; rs += p; }
#pragma unroll
      for (int m2 = 1; m2 < 16; m2 <<= 1) rs += __shfl_xor(rs, m2, 32);
      Lrun[r] = Lrun[r] * alpha + rs;
      Mrun[r] = nM;
#pragma unroll
      for (int t2 = 0; t2 < 16; ++t2) acc[t2][r] *= alpha;
    }

    // ---- P (C-layout) -> per-wave LDS -> A-layout fragments ----
#pragma unroll
    for (int j = 0; j < 4; ++j)
#pragma unroll
      for (int r = 0; r < 8; ++r)
        pT[w][half * 8 + r][j * 16 + l16] = f2bf(st[j][r]);

    // ---- O += P * V (16 x 256 per wave) ----
#pragma unroll
    for (int c2 = 0; c2 < 2; ++c2) {
      FragBF pa;
      const u16* pp = &pT[w][l16][c2 * 32];
      pa.q[0] = *(const uint4*)(pp + half * 8);
      pa.q[1] = *(const uint4*)(pp + 16 + half * 8);
#pragma unroll
      for (int t2 = 0; t2 < 16; ++t2) {
        FragBF vb;
        const u16* vp = &vT[t2 * 16 + l16][c2 * 32 + half * 8];
        vb.q[0] = *(const uint4*)(vp);
        vb.q[1] = *(const uint4*)(vp + 16);
        acc[t2] = __builtin_amdgcn_wmma_f32_16x16x32_bf16(
            false, pa.v, false, vb.v, (short)0, acc[t2], false, false);
      }
    }
  }

  // normalize and store bf16 (token-major [S, 4096])
  float inv[8];
#pragma unroll
  for (int r = 0; r < 8; ++r) inv[r] = 1.f / Lrun[r];
#pragma unroll
  for (int t2 = 0; t2 < 16; ++t2) {
    int col = (int)colQ + t2 * 16 + l16;
#pragma unroll
    for (int r = 0; r < 8; ++r)
      O[(size_t)(qRowBase + r) * D_MODEL + col] = f2bf(acc[t2][r] * inv[r]);
  }
}

// ---------------------------------------------------------------------------
// Host launcher
// ---------------------------------------------------------------------------
extern "C" void kernel_launch(void* const* d_in, const int* in_sizes, int n_in,
                              void* d_out, int out_size, void* d_ws, size_t ws_size,
                              hipStream_t stream) {
  (void)in_sizes; (void)n_in; (void)out_size; (void)ws_size;
  const float* X   = (const float*)d_in[0];
  const int*   pos = (const int*)d_in[1];
  const float* Wq  = (const float*)d_in[2];
  const float* Wk  = (const float*)d_in[3];
  const float* Wv  = (const float*)d_in[4];
  const float* Wo  = (const float*)d_in[5];
  float* out = (float*)d_out;

  const size_t SZX = (size_t)SEQ * D_MODEL;       //  8.4 M elems
  const size_t SZW = (size_t)D_MODEL * D_MODEL;   // 16.8 M elems
  u16* Xb  = (u16*)d_ws;                          // X            bf16 [S][D]
  u16* WqT = Xb  + SZX;                           // Wq^T         bf16 [D][D]
  u16* WkT = WqT + SZW;
  u16* WvT = WkT + SZW;
  u16* WoT = WvT + SZW;
  u16* Qb  = WoT + SZW;                           // Q,K,V,attnO  bf16 [S][D]
  u16* Kb  = Qb  + SZX;
  u16* Vb  = Kb  + SZX;
  u16* AOb = Vb  + SZX;                           // total ~208 MB of ws

  // 1) one-time precision/layout pass: X -> bf16; W -> bf16 transposed
  cvt_f32_bf16_k<<<dim3((unsigned)(SZX / 1024)), 256, 0, stream>>>(X, Xb, (int)SZX);
  dim3 tg(D_MODEL / 32, D_MODEL / 32);            // (128,128)
  cvtT_f32_bf16_k<<<tg, 256, 0, stream>>>(Wq, WqT, D_MODEL, D_MODEL);
  cvtT_f32_bf16_k<<<tg, 256, 0, stream>>>(Wk, WkT, D_MODEL, D_MODEL);
  cvtT_f32_bf16_k<<<tg, 256, 0, stream>>>(Wv, WvT, D_MODEL, D_MODEL);
  cvtT_f32_bf16_k<<<tg, 256, 0, stream>>>(Wo, WoT, D_MODEL, D_MODEL);

  // 2) QKV projections (LDS-free N-T bf16 WMMA GEMM, f32 accumulate)
  dim3 gg(D_MODEL / 128, SEQ / 128);              // (32, 16)
  gemm_nt_k<false><<<gg, 256, 0, stream>>>(Xb, WqT, Qb, SEQ, D_MODEL, D_MODEL);
  gemm_nt_k<false><<<gg, 256, 0, stream>>>(Xb, WkT, Kb, SEQ, D_MODEL, D_MODEL);
  gemm_nt_k<false><<<gg, 256, 0, stream>>>(Xb, WvT, Vb, SEQ, D_MODEL, D_MODEL);

  // 3) rotary embedding on Q, K
  rotary_k<<<dim3(SEQ * NHEAD * 32 / 256), 256, 0, stream>>>(Qb, Kb, pos);

  // 4) causal flash attention (async K staging + WMMA)
  flash_attn_k<<<dim3(SEQ / 64, NHEAD), 128, 0, stream>>>(Qb, Kb, Vb, AOb);

  // 5) output projection -> f32
  gemm_nt_k<true><<<gg, 256, 0, stream>>>(AOb, WoT, (void*)out, SEQ, D_MODEL, D_MODEL);
}
